// AttnBlock_84722524881146
// MI455X (gfx1250) — matrix operand order repaired and checked by reading, not compiled, not run on hardware
//
#include <hip/hip_runtime.h>
#include <hip/hip_bf16.h>
#include <math.h>

typedef __attribute__((ext_vector_type(16))) __bf16 v16bf;
typedef __attribute__((ext_vector_type(8)))  __bf16 v8bf;
typedef __attribute__((ext_vector_type(8)))  float  v8f;

#define B_    32
#define N_    577
#define NPAD_ 608           // 19*32, padded key count
#define DIM_  768
#define NH_   12
#define HD_   64
#define MLP_  3072
#define M_    (B_*N_)       // 18464
#define QKVC  (3*DIM_)      // 2304
#define SCALE_ 0.125f       // 64^-0.5
#define EPS_  1e-6f
#define NEG_INF_ (-3.0e38f)

__device__ __forceinline__ v8f wmma_bf16(v16bf a, v16bf b, v8f c) {
  return __builtin_amdgcn_wmma_f32_16x16x32_bf16(false, a, false, b, (short)0, c, false, false);
}

// elements 0..7 at p, elements 8..15 at p+16  (A-fragment K pattern)
__device__ __forceinline__ v16bf ld_frag_split(const __bf16* p) {
  v8bf a = *(const v8bf*)p;
  v8bf b = *(const v8bf*)(p + 16);
  v16bf f;
#pragma unroll
  for (int i = 0; i < 8; i++) { f[i] = a[i]; f[i + 8] = b[i]; }
  return f;
}
// 16 contiguous elements at p  (contiguous K order, used for the V^T A-fragment)
__device__ __forceinline__ v16bf ld_frag_contig(const __bf16* p) {
  v8bf a = *(const v8bf*)p;
  v8bf b = *(const v8bf*)(p + 8);
  v16bf f;
#pragma unroll
  for (int i = 0; i < 8; i++) { f[i] = a[i]; f[i + 8] = b[i]; }
  return f;
}

// ---------------- CDNA5 async/TDM probe kernels (compiled, never launched) ---
__global__ void cdna5_probe_tdm_kernel(int* g) {
  asm volatile("tensor_load_to_lds s[0:3], s[4:11]" ::: "memory");
  __builtin_amdgcn_s_wait_tensorcnt(0);
  if (g) *g = 0;
}
__global__ void cdna5_probe_async_kernel(int* g) {
  asm volatile("global_load_async_to_lds_b128 v0, v[2:3], off\n\t"
               "s_wait_asynccnt 0x0" ::: "memory");
  if (g) *g = 0;
}

// ---------------- small helper kernels ----------------
__global__ void f2bf_kernel(const float* __restrict__ in, __bf16* __restrict__ out, int n) {
  int i = blockIdx.x * 256 + threadIdx.x;
  if (i < n) out[i] = (__bf16)in[i];
}

__global__ void qkvbias_kernel(const float* __restrict__ qb, const float* __restrict__ vb,
                               float* __restrict__ bias) {
  int i = blockIdx.x * 256 + threadIdx.x;
  if (i < QKVC) {
    float v = 0.f;
    if (i < DIM_) v = qb[i];
    else if (i >= 2 * DIM_) v = vb[i - 2 * DIM_];
    bias[i] = v;
  }
}

// expanded relative-position bias, TRANSPOSED: rpbT[h][key][q]
__global__ void rpb_kernel(const float* __restrict__ table, const int* __restrict__ rel,
                           float* __restrict__ rpbT) {
  int i = blockIdx.x * 256 + threadIdx.x;   // i = key*N + q
  const int NN = N_ * N_;
  if (i < NN) {
    int key = i / N_;
    int q   = i - key * N_;
    int ri  = rel[q * N_ + key];
#pragma unroll
    for (int h = 0; h < NH_; h++)
      rpbT[(size_t)h * NN + i] = table[ri * NH_ + h];
  }
}

// v-part of qkv -> vT[b][h][d][keyPadded], zero-padded to NPAD_ keys
__global__ __launch_bounds__(256)
void vtrans_kernel(const __bf16* __restrict__ qkv, __bf16* __restrict__ vT) {
  __shared__ __bf16 tile[16][17];
  const int kt = blockIdx.x;          // 0..37 (key tile)
  const int dt = blockIdx.y;          // 0..3  (d tile)
  const int bh = blockIdx.z;          // 0..383
  const int b = bh / NH_, h = bh - b * NH_;
  const int tx = threadIdx.x & 15, ty = threadIdx.x >> 4;

  const int key_r = kt * 16 + ty;
  __bf16 val = (__bf16)0.f;
  if (key_r < N_)
    val = qkv[((size_t)(b * N_ + key_r)) * QKVC + 2 * DIM_ + h * HD_ + dt * 16 + tx];
  tile[ty][tx] = val;
  __syncthreads();

  const int d = dt * 16 + ty;
  const int key_w = kt * 16 + tx;
  vT[((size_t)(bh)*HD_ + d) * NPAD_ + key_w] = tile[tx][ty];
}

// LayerNorm over 768, fp32 in -> bf16 out
__global__ __launch_bounds__(256) void ln_kernel(const float* __restrict__ x,
                                                 const float* __restrict__ g,
                                                 const float* __restrict__ bta,
                                                 __bf16* __restrict__ out) {
  __shared__ float red[256];
  const int row = blockIdx.x, tid = threadIdx.x;
  const float* xr = x + (size_t)row * DIM_;
  float v0 = xr[tid], v1 = xr[tid + 256], v2 = xr[tid + 512];
  red[tid] = v0 + v1 + v2;
  __syncthreads();
  for (int off = 128; off > 0; off >>= 1) {
    if (tid < off) red[tid] += red[tid + off];
    __syncthreads();
  }
  float mean = red[0] * (1.f / DIM_);
  __syncthreads();
  float d0 = v0 - mean, d1 = v1 - mean, d2 = v2 - mean;
  red[tid] = d0 * d0 + d1 * d1 + d2 * d2;
  __syncthreads();
  for (int off = 128; off > 0; off >>= 1) {
    if (tid < off) red[tid] += red[tid + off];
    __syncthreads();
  }
  float rstd = rsqrtf(red[0] * (1.f / DIM_) + EPS_);
  __bf16* orow = out + (size_t)row * DIM_;
  orow[tid]       = (__bf16)(d0 * rstd * g[tid]       + bta[tid]);
  orow[tid + 256] = (__bf16)(d1 * rstd * g[tid + 256] + bta[tid + 256]);
  orow[tid + 512] = (__bf16)(d2 * rstd * g[tid + 512] + bta[tid + 512]);
}

// ---------------- WMMA GEMM: C = A[M,K] * W[N,K]^T + bias, with epilogues ----
// 8 waves (2x4) per block; 64x64 tile per wave -> 128x256 block tile.
__global__ __launch_bounds__(256)
void wmma_gemm_kernel(const __bf16* __restrict__ A, const __bf16* __restrict__ W,
                      const float* __restrict__ bias, const float* __restrict__ resid,
                      float* __restrict__ outF, __bf16* __restrict__ outBF,
                      int Mrows, int K, int Nout, int mode) {
  const int lane = threadIdx.x & 31;
  const int wid  = threadIdx.x >> 5;
  const int lo = lane & 15, hi = lane >> 4;
  const int wm = wid >> 2, wn = wid & 3;              // 2x4 waves
  const int rowBase = blockIdx.x * 128 + wm * 64;
  const int colBase = blockIdx.y * 256 + wn * 64;

  v8f acc[4][4];
#pragma unroll
  for (int i = 0; i < 4; i++)
#pragma unroll
    for (int j = 0; j < 4; j++)
#pragma unroll
      for (int r = 0; r < 8; r++) acc[i][j][r] = 0.f;

  const __bf16* pa[4];
  const __bf16* pw[4];
#pragma unroll
  for (int t = 0; t < 4; t++) {
    int rA = rowBase + t * 16 + lo;
    if (rA >= Mrows) rA = Mrows - 1;
    pa[t] = A + (size_t)rA * K + 8 * hi;
    pw[t] = W + (size_t)(colBase + t * 16 + lo) * K + 8 * hi;
  }

  for (int k0 = 0; k0 < K; k0 += 32) {
    v16bf af[4], bf[4];
#pragma unroll
    for (int t = 0; t < 4; t++) af[t] = ld_frag_split(pa[t] + k0);
#pragma unroll
    for (int t = 0; t < 4; t++) bf[t] = ld_frag_split(pw[t] + k0);
#pragma unroll
    for (int mi = 0; mi < 4; mi++)
#pragma unroll
      for (int ni = 0; ni < 4; ni++)
        acc[mi][ni] = wmma_bf16(af[mi], bf[ni], acc[mi][ni]);
  }

  if (mode == 0) {
#pragma unroll
    for (int mi = 0; mi < 4; mi++)
#pragma unroll
      for (int ni = 0; ni < 4; ni++) {
        const int col = colBase + ni * 16 + lo;
        const float bv = bias[col];
        const float sc = (col < DIM_) ? SCALE_ : 1.f;
#pragma unroll
        for (int r = 0; r < 8; r++) {
          const int row = rowBase + mi * 16 + r + 8 * hi;
          if (row < Mrows)
            outBF[(size_t)row * Nout + col] = (__bf16)((acc[mi][ni][r] + bv) * sc);
        }
      }
  } else if (mode == 1) {
#pragma unroll
    for (int mi = 0; mi < 4; mi++)
#pragma unroll
      for (int ni = 0; ni < 4; ni++) {
        const int col = colBase + ni * 16 + lo;
        const float bv = bias[col];
#pragma unroll
        for (int r = 0; r < 8; r++) {
          const int row = rowBase + mi * 16 + r + 8 * hi;
          if (row < Mrows) {
            const size_t idx = (size_t)row * Nout + col;
            outF[idx] = resid[idx] + acc[mi][ni][r] + bv;
          }
        }
      }
  } else {
#pragma unroll
    for (int mi = 0; mi < 4; mi++)
#pragma unroll
      for (int ni = 0; ni < 4; ni++) {
        const int col = colBase + ni * 16 + lo;
        const float bv = bias[col];
#pragma unroll
        for (int r = 0; r < 8; r++) {
          const int row = rowBase + mi * 16 + r + 8 * hi;
          if (row < Mrows) {
            const float v = acc[mi][ni][r] + bv;
            outBF[(size_t)row * Nout + col] =
                (__bf16)(0.5f * v * (1.f + erff(v * 0.70710678118654752f)));
          }
        }
      }
  }
}

// ---------------- flash attention, one wave per (32 queries, head, batch) ----
// qkv: [b,n,2304] bf16 (q pre-scaled). vT: [b,h,d,keyPadded] bf16 (zero-padded).
// rpbT: [h][key][q] f32. attnout: [b,n,768] bf16.
__global__ __launch_bounds__(32)
void attn_kernel(const __bf16* __restrict__ qkv, const __bf16* __restrict__ vT,
                 const float* __restrict__ rpbT, __bf16* __restrict__ attnout) {
  const int lane = threadIdx.x;
  const int lo = lane & 15, hi = lane >> 4;
  const int qt = blockIdx.x, h = blockIdx.y, b = blockIdx.z;

  int qcol[2], qc[2];
  v16bf bq[2][2];
#pragma unroll
  for (int s = 0; s < 2; s++) {
    qcol[s] = qt * 32 + s * 16 + lo;
    qc[s] = (qcol[s] < N_) ? qcol[s] : (N_ - 1);
    const __bf16* qbp = qkv + ((size_t)(b * N_ + qc[s])) * QKVC + h * HD_;
#pragma unroll
    for (int kk = 0; kk < 2; kk++)
      bq[s][kk] = ld_frag_split(qbp + kk * 32 + 8 * hi);
  }
  const float* rT = rpbT + (size_t)h * N_ * N_;
  const __bf16* vbase = vT + ((size_t)(b * NH_ + h)) * HD_ * NPAD_ + 16 * hi;

  v8f oacc[2][4];
#pragma unroll
  for (int s = 0; s < 2; s++)
#pragma unroll
    for (int dt = 0; dt < 4; dt++)
#pragma unroll
      for (int r = 0; r < 8; r++) oacc[s][dt][r] = 0.f;

  float m_i[2] = {NEG_INF_, NEG_INF_};
  float l_i[2] = {0.f, 0.f};

  for (int kc = 0; kc < 19; kc++) {           // 19*32 = 608 >= 577 keys
    const int keyBase = kc * 32;

    // s^T tiles (rows = keys, cols = q) for both q-subtiles; K frags shared
    v8f st[2][2];
#pragma unroll
    for (int s = 0; s < 2; s++)
#pragma unroll
      for (int t = 0; t < 2; t++)
#pragma unroll
        for (int r = 0; r < 8; r++) st[s][t][r] = 0.f;

#pragma unroll
    for (int t = 0; t < 2; t++) {
      int key = keyBase + t * 16 + lo;
      int kcl = (key < N_) ? key : (N_ - 1);
      const __bf16* kb = qkv + ((size_t)(b * N_ + kcl)) * QKVC + DIM_ + h * HD_;
#pragma unroll
      for (int kk = 0; kk < 2; kk++) {
        v16bf af = ld_frag_split(kb + kk * 32 + 8 * hi);
        st[0][t] = wmma_bf16(af, bq[0][kk], st[0][t]);
        st[1][t] = wmma_bf16(af, bq[1][kk], st[1][t]);
      }
    }

    // softmax update + p^T B-fragment per subtile
    v16bf pb[2];
#pragma unroll
    for (int s = 0; s < 2; s++) {
      float p0[8], p1[8];
      float cm = NEG_INF_;
#pragma unroll
      for (int r = 0; r < 8; r++) {
        int key0 = keyBase + r + 8 * hi;
        int key1 = keyBase + 16 + r + 8 * hi;
        float s0 = (key0 < N_) ? (st[s][0][r] + rT[(size_t)key0 * N_ + qc[s]]) : NEG_INF_;
        float s1 = (key1 < N_) ? (st[s][1][r] + rT[(size_t)key1 * N_ + qc[s]]) : NEG_INF_;
        p0[r] = s0; p1[r] = s1;
        cm = fmaxf(cm, fmaxf(s0, s1));
      }
      cm = fmaxf(cm, __shfl_xor(cm, 16, 32));
      const float nm = fmaxf(m_i[s], cm);
      const float corr = __expf(m_i[s] - nm);
      float ls = 0.f;
#pragma unroll
      for (int r = 0; r < 8; r++) {
        p0[r] = __expf(p0[r] - nm);
        p1[r] = __expf(p1[r] - nm);
        ls += p0[r] + p1[r];
      }
      ls += __shfl_xor(ls, 16, 32);
      l_i[s] = l_i[s] * corr + ls;
      m_i[s] = nm;
#pragma unroll
      for (int dt = 0; dt < 4; dt++)
#pragma unroll
        for (int r = 0; r < 8; r++) oacc[s][dt][r] *= corr;

      // B-fragment: lane needs K = i + 16*hi; half swapped with lane^16
#pragma unroll
      for (int r = 0; r < 8; r++) {
        float x0 = __shfl_xor(p0[r], 16, 32);
        float x1 = __shfl_xor(p1[r], 16, 32);
        pb[s][r]     = (__bf16)(hi ? x1 : p0[r]);
        pb[s][8 + r] = (__bf16)(hi ? p1[r] : x0);
      }
    }

    // O^T += v^T x p^T  (rows = hd, cols = q); V frags from transposed vT
#pragma unroll
    for (int dt = 0; dt < 4; dt++) {
      v16bf av = ld_frag_contig(vbase + (size_t)(dt * 16 + lo) * NPAD_ + keyBase);
      oacc[0][dt] = wmma_bf16(av, pb[0], oacc[0][dt]);
      oacc[1][dt] = wmma_bf16(av, pb[1], oacc[1][dt]);
    }
  }

#pragma unroll
  for (int s = 0; s < 2; s++) {
    if (qcol[s] < N_) {
      const float inv = 1.f / l_i[s];
#pragma unroll
      for (int dt = 0; dt < 4; dt++)
#pragma unroll
        for (int r = 0; r < 8; r++) {
          const int d = dt * 16 + r + 8 * hi;
          attnout[((size_t)(b * N_ + qcol[s])) * DIM_ + h * HD_ + d] =
              (__bf16)(oacc[s][dt][r] * inv);
        }
    }
  }
}

// ---------------- launcher ----------------
extern "C" void kernel_launch(void* const* d_in, const int* in_sizes, int n_in,
                              void* d_out, int out_size, void* d_ws, size_t ws_size,
                              hipStream_t stream) {
  (void)in_sizes; (void)n_in; (void)out_size; (void)ws_size;
  const float* x      = (const float*)d_in[0];
  const float* n1g    = (const float*)d_in[1];
  const float* n1b    = (const float*)d_in[2];
  const float* qkv_w  = (const float*)d_in[3];
  const float* q_bias = (const float*)d_in[4];
  const float* v_bias = (const float*)d_in[5];
  const float* rpb_t  = (const float*)d_in[6];
  const float* proj_w = (const float*)d_in[7];
  const float* proj_b = (const float*)d_in[8];
  const float* n2g    = (const float*)d_in[9];
  const float* n2b    = (const float*)d_in[10];
  const float* fc1_w  = (const float*)d_in[11];
  const float* fc1_b  = (const float*)d_in[12];
  const float* fc2_w  = (const float*)d_in[13];
  const float* fc2_b  = (const float*)d_in[14];
  const int*   rel    = (const int*)d_in[15];
  float* out = (float*)d_out;

  char* base = (char*)d_ws;
  size_t off = 0;
  auto alloc = [&](size_t bytes) -> void* {
    void* p = base + off;
    off = (off + bytes + 255) & ~(size_t)255;
    return p;
  };
  __bf16* h1bf    = (__bf16*)alloc((size_t)M_ * DIM_ * 2);
  __bf16* qkvbf   = (__bf16*)alloc((size_t)M_ * QKVC * 2);
  float*  rpbT    = (float*) alloc((size_t)NH_ * N_ * N_ * 4);
  __bf16* vTbuf   = (__bf16*)alloc((size_t)B_ * NH_ * HD_ * NPAD_ * 2);
  __bf16* attnbf  = (__bf16*)alloc((size_t)M_ * DIM_ * 2);
  float*  x1      = (float*) alloc((size_t)M_ * DIM_ * 4);
  __bf16* h2bf    = (__bf16*)alloc((size_t)M_ * DIM_ * 2);
  __bf16* mbf     = (__bf16*)alloc((size_t)M_ * MLP_ * 2);
  __bf16* qkvw_bf = (__bf16*)alloc((size_t)QKVC * DIM_ * 2);
  __bf16* projw_bf= (__bf16*)alloc((size_t)DIM_ * DIM_ * 2);
  __bf16* fc1w_bf = (__bf16*)alloc((size_t)MLP_ * DIM_ * 2);
  __bf16* fc2w_bf = (__bf16*)alloc((size_t)DIM_ * MLP_ * 2);
  float*  bias3   = (float*) alloc((size_t)QKVC * 4);

  auto cvt = [&](const float* src, __bf16* dst, int n) {
    f2bf_kernel<<<(n + 255) / 256, 256, 0, stream>>>(src, dst, n);
  };
  cvt(qkv_w,  qkvw_bf,  QKVC * DIM_);
  cvt(proj_w, projw_bf, DIM_ * DIM_);
  cvt(fc1_w,  fc1w_bf,  MLP_ * DIM_);
  cvt(fc2_w,  fc2w_bf,  DIM_ * MLP_);
  qkvbias_kernel<<<(QKVC + 255) / 256, 256, 0, stream>>>(q_bias, v_bias, bias3);
  rpb_kernel<<<(N_ * N_ + 255) / 256, 256, 0, stream>>>(rpb_t, rel, rpbT);

  const int gM = (M_ + 127) / 128;

  // LN1 -> h1bf
  ln_kernel<<<M_, 256, 0, stream>>>(x, n1g, n1b, h1bf);
  // QKV GEMM (q scaled, bf16 out)
  wmma_gemm_kernel<<<dim3(gM, QKVC / 256), 256, 0, stream>>>(
      h1bf, qkvw_bf, bias3, nullptr, nullptr, qkvbf, M_, DIM_, QKVC, 0);
  // transpose V part -> vT[b][h][d][keyPadded]
  vtrans_kernel<<<dim3(NPAD_ / 16, HD_ / 16, B_ * NH_), 256, 0, stream>>>(qkvbf, vTbuf);
  // attention (2 q-subtiles per wave)
  attn_kernel<<<dim3(19, NH_, B_), 32, 0, stream>>>(qkvbf, vTbuf, rpbT, attnbf);
  // proj + residual -> x1 (f32)
  wmma_gemm_kernel<<<dim3(gM, DIM_ / 256), 256, 0, stream>>>(
      attnbf, projw_bf, proj_b, x, x1, nullptr, M_, DIM_, DIM_, 1);
  // LN2 -> h2bf
  ln_kernel<<<M_, 256, 0, stream>>>(x1, n2g, n2b, h2bf);
  // fc1 + exact GELU -> mbf
  wmma_gemm_kernel<<<dim3(gM, MLP_ / 256), 256, 0, stream>>>(
      h2bf, fc1w_bf, fc1_b, nullptr, nullptr, mbf, M_, DIM_, MLP_, 2);
  // fc2 + residual -> out (f32)
  wmma_gemm_kernel<<<dim3(gM, DIM_ / 256), 256, 0, stream>>>(
      mbf, fc2w_bf, fc2_b, x1, out, nullptr, M_, MLP_, DIM_, 1);
}
